// DeepSeekMoE_39530878992791
// MI455X (gfx1250) — compile-verified, matrix operand order checked by reading, not run on hardware
//
#include <hip/hip_runtime.h>

// ---------------- types ----------------
typedef __attribute__((ext_vector_type(16))) __bf16 v16bf;
typedef __attribute__((ext_vector_type(8)))  __bf16 v8bf;
typedef __attribute__((ext_vector_type(8)))  float  v8f;
typedef __attribute__((ext_vector_type(4)))  unsigned int v4u;
typedef __attribute__((ext_vector_type(8)))  int v8i;
typedef __attribute__((ext_vector_type(4)))  int v4i;

// ---------------- problem dims ----------------
constexpr int CD  = 1024;   // model dim C
constexpr int HD  = 4096;   // mlp hidden H
constexpr int ES  = 2;      // shared experts
constexpr int ER  = 8;      // routed experts
constexpr int TOK = 2048;   // B*T
constexpr int CAP = 2048;   // per-expert slot capacity

// ---------------- GEMM tiling ----------------
constexpr int TM  = 64;
constexpr int TN  = 128;
constexpr int TK  = 32;
constexpr int LDT = 40;     // padded LDS row stride (elements); 80B = TDM 64B row + 16B pad

__device__ __forceinline__ float gelu_exact(float x){
  return 0.5f * x * (1.0f + erff(x * 0.70710678118654752440f));
}

// One staged K-slab of WMMA: each wave owns 16 rows x 64 cols (4 x 16x16 frags).
__device__ __forceinline__ void mma_tile(const __bf16* As, const __bf16* Bs,
                                         int lane, int wm, int wn, v8f* acc){
  const int lm = lane & 15;
  const int kb = (lane >> 4) << 3;          // 0 or 8
  v16bf a;
  {
    const __bf16* p = As + (wm + lm) * LDT + kb;
    v8bf a0 = *(const v8bf*)p;
    v8bf a1 = *(const v8bf*)(p + 16);
    #pragma unroll
    for (int i = 0; i < 8; ++i){ a[i] = a0[i]; a[8 + i] = a1[i]; }
  }
  #pragma unroll
  for (int j = 0; j < 4; ++j){
    const __bf16* q = Bs + (wn + j * 16 + lm) * LDT + kb;
    v8bf b0 = *(const v8bf*)q;
    v8bf b1 = *(const v8bf*)(q + 16);
    v16bf b;
    #pragma unroll
    for (int i = 0; i < 8; ++i){ b[i] = b0[i]; b[8 + i] = b1[i]; }
    acc[j] = __builtin_amdgcn_wmma_f32_16x16x32_bf16(false, a, false, b,
                                                     (short)0, acc[j], false, false);
  }
}

// LDS byte offset of a generic pointer to __shared__ (aperture keeps the
// workgroup-relative offset in addr[31:0]).
__device__ __forceinline__ unsigned lds_off(const void* p){
  return (unsigned)(unsigned long long)(uintptr_t)p;
}

// Per-thread A-slab async copy (1 x b128 per thread), ASYNCcnt-tracked.
__device__ __forceinline__ void async_stage_A(const __bf16* aRow, int k0,
                                              __bf16* AsB, int ar, int ac){
  unsigned la = lds_off(AsB + ar * LDT + ac);
  unsigned long long ga = (unsigned long long)(uintptr_t)(aRow + k0);
  asm volatile("global_load_async_to_lds_b128 %0, %1, off"
               :: "v"(la), "v"(ga) : "memory");
}

// Tensor Data Mover: DMA one 128x32 bf16 weight tile (row stride Kstride elems)
// into LDS with hardware padding to the LDT=40 row pitch. One issue per block.
// D# layout per CDNA5 ISA 8.3/8.4 (groups 2/3 NULL -> 2D tensor).
// This toolchain exposes the 6-arg builtin: (g0 u32x4, g1 i32x8, g2 i32x4,
// g3 i32x4, i32x8, cpol).
__device__ __forceinline__ void tdm_load_B(const __bf16* gsrc, __bf16* ldst, int Kstride){
  unsigned long long ga = (unsigned long long)(uintptr_t)gsrc;
  v4u g0;
  g0[0] = 1u;                                        // count=1 (valid), user mode
  g0[1] = lds_off(ldst);                             // lds_addr
  g0[2] = (unsigned)(ga & 0xFFFFFFFFu);              // global_addr[31:0]
  g0[3] = (unsigned)((ga >> 32) & 0x1FFFFFFu)        // global_addr[56:32]
        | (2u << 30);                                // type=2 ("image")
  v8i g1;
  g1[0] = (int)((1u << 16)                           // data_size = 2B
              | (1u << 20)                           // pad_enable
              | (3u << 22)                           // pad_interval: 16 DWORDs (64B row)
              | (3u << 25));                         // pad_amount: 4 DWORDs (16B)
  g1[1] = (int)((unsigned)TK << 16);                 // tensor_dim0 = 32 (low16 @ bits 63:48)
  g1[2] = (int)((unsigned)TN << 16);                 // dim0 hi=0 | tensor_dim1=128 (low16)
  g1[3] = (int)((unsigned)TK << 16);                 // dim1 hi=0 | tile_dim0 = 32
  g1[4] = TN;                                        // tile_dim1 = 128, tile_dim2 = 0
  g1[5] = Kstride;                                   // tensor_dim0_stride[31:0]
  g1[6] = 0;                                         // stride hi | tensor_dim1_stride lo
  g1[7] = 0;
  v4i z4 = {0, 0, 0, 0};
  v8i z8 = {0, 0, 0, 0, 0, 0, 0, 0};
  __builtin_amdgcn_tensor_load_to_lds(g0, g1, z4, z4, z8, 0);
}

// Double-buffered pipeline: TDM moves B tiles, per-lane async moves (gathered)
// A tiles; both overlap the WMMA stream one stage deep.
__device__ __forceinline__ void gemm_pipeline(const __bf16* aRow, const __bf16* bTile,
                                              int Kstride, __bf16* As, __bf16* Bs,
                                              int steps, int ar, int ac,
                                              int wave, int lane, int wm, int wn, v8f* acc){
  if (wave == 0) tdm_load_B(bTile, Bs, Kstride);
  async_stage_A(aRow, 0, As, ar, ac);
  for (int s = 0; s < steps; ++s){
    __bf16* cA = As + (s & 1) * (TM * LDT);
    __bf16* cB = Bs + (s & 1) * (TN * LDT);
    __syncthreads();   // all waves done reading buf[(s+1)&1] (their s-1 MMA)
    if (s + 1 < steps){
      if (wave == 0)
        tdm_load_B(bTile + (s + 1) * TK, Bs + ((s + 1) & 1) * (TN * LDT), Kstride);
      async_stage_A(aRow, (s + 1) * TK, As + ((s + 1) & 1) * (TM * LDT), ar, ac);
      __builtin_prefetch(aRow + (s + 2) * TK, 0, 2);
      // in-order retirement per counter: <=1 outstanding => stage s landed
      asm volatile("s_wait_asynccnt 0x1" ::: "memory");
      if (wave == 0) __builtin_amdgcn_s_wait_tensorcnt(1);
    } else {
      asm volatile("s_wait_asynccnt 0x0" ::: "memory");
      if (wave == 0) __builtin_amdgcn_s_wait_tensorcnt(0);
    }
    __syncthreads();   // publish stage-s LDS data to every wave
    mma_tile(cA, cB, lane, wm, wn, acc);
  }
}

// ---------------- kernels ----------------

__global__ void k_zero_counts(int* counts){
  if (threadIdx.x < ER) counts[threadIdx.x] = 0;
}

// fp32 [E][K][N] -> bf16 [E][N][K] (K-contiguous rows for WMMA B fragments)
__global__ void __launch_bounds__(256) k_convert(const float* __restrict__ src,
                                                 __bf16* __restrict__ dst,
                                                 int E, int Kd, int Nd){
  long long idx = (long long)blockIdx.x * blockDim.x + threadIdx.x;
  long long total = (long long)E * Kd * Nd;
  if (idx >= total) return;
  int n = (int)(idx % Nd);
  long long t = idx / Nd;
  int k = (int)(t % Kd);
  int e = (int)(t / Kd);
  dst[((long long)e * Nd + n) * Kd + k] = (__bf16)src[idx];
}

// RMSNorm (both gains) to bf16 + sigmoid top-2 routing; builds expert lists.
__global__ void __launch_bounds__(256)
k_norm_route(const float* __restrict__ u, const float* __restrict__ gs,
             const float* __restrict__ gr, const float* __restrict__ cent,
             __bf16* __restrict__ Xs, __bf16* __restrict__ Xr,
             int* __restrict__ counts, int* __restrict__ tokl,
             int* __restrict__ slotof, float* __restrict__ gateof)
{
  const int t = blockIdx.x;
  const float* x = u + (size_t)t * CD;

  float ss = 0.f;
  float sc[ER];
  #pragma unroll
  for (int e = 0; e < ER; ++e) sc[e] = 0.f;

  for (int c = threadIdx.x; c < CD; c += 256){
    float v = x[c];
    ss += v * v;
    const float* cr = cent + (size_t)c * ER;   // centroids [C][ER]
    #pragma unroll
    for (int e = 0; e < ER; ++e) sc[e] += v * cr[e];
  }
  // wave32 reduction
  #pragma unroll
  for (int off = 16; off > 0; off >>= 1){
    ss += __shfl_down(ss, off, 32);
    #pragma unroll
    for (int e = 0; e < ER; ++e) sc[e] += __shfl_down(sc[e], off, 32);
  }

  __shared__ float red[8][ER + 1];
  __shared__ float rnorm_s;
  const int wv = threadIdx.x >> 5, ln = threadIdx.x & 31;
  if (ln == 0){
    red[wv][0] = ss;
    #pragma unroll
    for (int e = 0; e < ER; ++e) red[wv][1 + e] = sc[e];
  }
  __syncthreads();

  if (threadIdx.x == 0){
    float S = 0.f;
    float sce[ER];
    #pragma unroll
    for (int e = 0; e < ER; ++e) sce[e] = 0.f;
    for (int w2 = 0; w2 < 8; ++w2){
      S += red[w2][0];
      #pragma unroll
      for (int e = 0; e < ER; ++e) sce[e] += red[w2][1 + e];
    }
    rnorm_s = rsqrtf(S * (1.0f / CD) + 1.1920929e-07f);   // eps = finfo(f32).eps

    float sig[ER];
    float sum = 0.f;
    #pragma unroll
    for (int e = 0; e < ER; ++e){ sig[e] = 1.0f / (1.0f + expf(-sce[e])); sum += sig[e]; }
    // top-2, first-index-wins on ties (matches lax.top_k)
    int i0 = 0;
    for (int e = 1; e < ER; ++e) if (sig[e] > sig[i0]) i0 = e;
    int i1 = (i0 == 0) ? 1 : 0;
    for (int e = 0; e < ER; ++e) if (e != i0 && sig[e] > sig[i1]) i1 = e;
    const float inv = 1.0f / sum;
    int   idx[2] = { i0, i1 };
    float gv[2]  = { sig[i0] * inv, sig[i1] * inv };
    #pragma unroll
    for (int j = 0; j < 2; ++j){
      int p = atomicAdd(&counts[idx[j]], 1);
      int slot = idx[j] * CAP + p;
      tokl[slot] = t;
      slotof[t * 2 + j] = slot;
      gateof[t * 2 + j] = gv[j];
    }
  }
  __syncthreads();

  const float rn = rnorm_s;
  for (int c = threadIdx.x; c < CD; c += 256){
    float v = x[c] * rn;
    Xs[(size_t)t * CD + c] = (__bf16)(v * gs[c]);
    Xr[(size_t)t * CD + c] = (__bf16)(v * gr[c]);
  }
}

// GEMM1: H[slot][HD] = gelu(X[token] * W1t[e]^T + b1[e]); optional token gather.
__global__ void __launch_bounds__(256)
k_gemm_h(const __bf16* __restrict__ X, const __bf16* __restrict__ Wt,
         const float* __restrict__ b1, __bf16* __restrict__ Hout,
         const int* __restrict__ tokl, const int* __restrict__ counts,
         int use_list)
{
  const int e  = blockIdx.z;
  const int m0 = blockIdx.y * TM;
  const int n0 = blockIdx.x * TN;
  const int count = use_list ? counts[e] : TOK;
  if (m0 >= count) return;

  __shared__ __align__(16) __bf16 As[2 * TM * LDT];
  __shared__ __align__(16) __bf16 Bs[2 * TN * LDT];
  __shared__ int tok_s[TM];

  const int tid = threadIdx.x;
  if (tid < TM){
    int r  = m0 + tid;
    int rc = (r < count) ? r : (count - 1);
    tok_s[tid] = use_list ? tokl[e * CAP + rc] : rc;
  }
  __syncthreads();

  const __bf16* We = Wt + (size_t)e * HD * CD;   // [HD][CD] k-contiguous
  const float*  be = b1 + (size_t)e * HD;

  const int ar = tid >> 2;            // A row 0..63
  const int ac = (tid & 3) * 8;       // A k chunk

  const __bf16* aRow  = X + (size_t)tok_s[ar] * CD + ac;
  const __bf16* bTile = We + (size_t)n0 * CD;    // uniform tile origin for TDM

  const int wave = tid >> 5, lane = tid & 31;
  const int wm = (wave & 3) * 16;
  const int wn = (wave >> 2) * 64;

  v8f acc[4];
  #pragma unroll
  for (int j = 0; j < 4; ++j)
    #pragma unroll
    for (int i = 0; i < 8; ++i) acc[j][i] = 0.f;

  gemm_pipeline(aRow, bTile, CD, As, Bs, CD / TK, ar, ac, wave, lane, wm, wn, acc);

  const int hi = (lane >> 4) << 3;    // C-frag: lanes 16..31 hold rows r+8
  #pragma unroll
  for (int j = 0; j < 4; ++j){
    const int n = n0 + wn + j * 16 + (lane & 15);
    const float bias = be[n];
    #pragma unroll
    for (int r = 0; r < 8; ++r){
      const int m = m0 + wm + r + hi;
      float v = gelu_exact(acc[j][r] + bias);
      Hout[((size_t)(e * CAP + m)) * HD + n] = (__bf16)v;
    }
  }
}

// GEMM2 shared: out[t][c] = u + sum_e( Hs[e][t] * W2t[e]^T + b2[e] ). Fully writes d_out.
__global__ void __launch_bounds__(256)
k_out_shared(const __bf16* __restrict__ Hs, const __bf16* __restrict__ W2t,
             const float* __restrict__ b2, const float* __restrict__ u,
             float* __restrict__ out)
{
  const int m0 = blockIdx.y * TM;
  const int n0 = blockIdx.x * TN;

  __shared__ __align__(16) __bf16 As[2 * TM * LDT];
  __shared__ __align__(16) __bf16 Bs[2 * TN * LDT];

  const int tid = threadIdx.x;
  const int ar = tid >> 2, ac = (tid & 3) * 8;
  const int wave = tid >> 5, lane = tid & 31;
  const int wm = (wave & 3) * 16, wn = (wave >> 2) * 64;

  v8f acc[4];
  #pragma unroll
  for (int j = 0; j < 4; ++j)
    #pragma unroll
    for (int i = 0; i < 8; ++i) acc[j][i] = 0.f;

  for (int e = 0; e < ES; ++e){
    const __bf16* aRow  = Hs  + ((size_t)(e * CAP + m0 + ar)) * HD + ac;
    const __bf16* bTile = W2t + ((size_t)(e * CD + n0)) * HD;
    gemm_pipeline(aRow, bTile, HD, As, Bs, HD / TK, ar, ac, wave, lane, wm, wn, acc);
  }

  const int hi = (lane >> 4) << 3;
  #pragma unroll
  for (int j = 0; j < 4; ++j){
    const int n = n0 + wn + j * 16 + (lane & 15);
    float bsum = 0.f;
    #pragma unroll
    for (int e = 0; e < ES; ++e) bsum += b2[e * CD + n];
    #pragma unroll
    for (int r = 0; r < 8; ++r){
      const int m = m0 + wm + r + hi;
      const size_t o = (size_t)m * CD + n;
      out[o] = u[o] + acc[j][r] + bsum;
    }
  }
}

// GEMM2 routed: Or[slot][c] = Hr[slot] * W2t[e]^T + b2[e]  (gate applied at finalize)
__global__ void __launch_bounds__(256)
k_out_routed(const __bf16* __restrict__ Hr, const __bf16* __restrict__ W2t,
             const float* __restrict__ b2, const int* __restrict__ counts,
             float* __restrict__ Or)
{
  const int e  = blockIdx.z;
  const int m0 = blockIdx.y * TM;
  const int n0 = blockIdx.x * TN;
  if (m0 >= counts[e]) return;

  __shared__ __align__(16) __bf16 As[2 * TM * LDT];
  __shared__ __align__(16) __bf16 Bs[2 * TN * LDT];

  const int tid = threadIdx.x;
  const int ar = tid >> 2, ac = (tid & 3) * 8;
  const int wave = tid >> 5, lane = tid & 31;
  const int wm = (wave & 3) * 16, wn = (wave >> 2) * 64;

  const __bf16* aRow  = Hr  + ((size_t)(e * CAP + m0 + ar)) * HD + ac;
  const __bf16* bTile = W2t + ((size_t)(e * CD + n0)) * HD;

  v8f acc[4];
  #pragma unroll
  for (int j = 0; j < 4; ++j)
    #pragma unroll
    for (int i = 0; i < 8; ++i) acc[j][i] = 0.f;

  gemm_pipeline(aRow, bTile, HD, As, Bs, HD / TK, ar, ac, wave, lane, wm, wn, acc);

  const int hi = (lane >> 4) << 3;
  #pragma unroll
  for (int j = 0; j < 4; ++j){
    const int n = n0 + wn + j * 16 + (lane & 15);
    const float bias = b2[e * CD + n];
    #pragma unroll
    for (int r = 0; r < 8; ++r){
      const int m = m0 + wm + r + hi;
      Or[((size_t)(e * CAP + m)) * CD + n] = acc[j][r] + bias;
    }
  }
}

// Deterministic combine: out[t] += g0*Or[slot0] + g1*Or[slot1]
__global__ void __launch_bounds__(256)
k_finalize(const float* __restrict__ Or, const int* __restrict__ slotof,
           const float* __restrict__ gateof, float* __restrict__ out)
{
  const int t = blockIdx.x;
  const int s0 = slotof[t * 2 + 0], s1 = slotof[t * 2 + 1];
  const float g0 = gateof[t * 2 + 0], g1 = gateof[t * 2 + 1];
  const float* r0 = Or + (size_t)s0 * CD;
  const float* r1 = Or + (size_t)s1 * CD;
  float* o = out + (size_t)t * CD;
  for (int c = threadIdx.x; c < CD; c += 256)
    o[c] += g0 * r0[c] + g1 * r1[c];
}

// ---------------- launcher ----------------
extern "C" void kernel_launch(void* const* d_in, const int* in_sizes, int n_in,
                              void* d_out, int out_size, void* d_ws, size_t ws_size,
                              hipStream_t stream) {
  const float* u     = (const float*)d_in[0];
  const float* g_s   = (const float*)d_in[1];
  const float* W1_s  = (const float*)d_in[2];
  const float* b1_s  = (const float*)d_in[3];
  const float* W2_s  = (const float*)d_in[4];
  const float* b2_s  = (const float*)d_in[5];
  const float* g_r   = (const float*)d_in[6];
  const float* W1_r  = (const float*)d_in[7];
  const float* b1_r  = (const float*)d_in[8];
  const float* W2_r  = (const float*)d_in[9];
  const float* b2_r  = (const float*)d_in[10];
  const float* cent  = (const float*)d_in[11];
  float* out = (float*)d_out;

  // workspace carve-up (256B aligned chunks)
  char* w = (char*)d_ws;
  size_t o = 0;
  auto carve = [&](size_t bytes) -> void* {
    void* p = w + o;
    o += (bytes + 255) & ~(size_t)255;
    return p;
  };
  int*    counts = (int*)   carve((size_t)ER * sizeof(int));
  __bf16* W1ts   = (__bf16*)carve((size_t)ES * HD * CD * 2);   // [e][HD][CD]
  __bf16* W2ts   = (__bf16*)carve((size_t)ES * CD * HD * 2);   // [e][CD][HD]
  __bf16* W1tr   = (__bf16*)carve((size_t)ER * HD * CD * 2);
  __bf16* W2tr   = (__bf16*)carve((size_t)ER * CD * HD * 2);
  __bf16* Xs     = (__bf16*)carve((size_t)TOK * CD * 2);
  __bf16* Xr     = (__bf16*)carve((size_t)TOK * CD * 2);
  __bf16* Hs     = (__bf16*)carve((size_t)ES * CAP * HD * 2);
  __bf16* Hr     = (__bf16*)carve((size_t)ER * CAP * HD * 2);
  float*  Or     = (float*) carve((size_t)ER * CAP * CD * 4);
  int*    tokl   = (int*)   carve((size_t)ER * CAP * 4);
  int*    slotof = (int*)   carve((size_t)TOK * 2 * 4);
  float*  gateof = (float*) carve((size_t)TOK * 2 * 4);

  k_zero_counts<<<1, 64, 0, stream>>>(counts);

  auto conv = [&](const float* src, __bf16* dst, int E, int Kd, int Nd){
    long long total = (long long)E * Kd * Nd;
    int blocks = (int)((total + 255) / 256);
    k_convert<<<blocks, 256, 0, stream>>>(src, dst, E, Kd, Nd);
  };
  conv(W1_s, W1ts, ES, CD, HD);   // [e][C][H] -> [e][H][C]
  conv(W2_s, W2ts, ES, HD, CD);   // [e][H][C] -> [e][C][H]
  conv(W1_r, W1tr, ER, CD, HD);
  conv(W2_r, W2tr, ER, HD, CD);

  k_norm_route<<<TOK, 256, 0, stream>>>(u, g_s, g_r, cent, Xs, Xr,
                                        counts, tokl, slotof, gateof);

  // GEMM1: shared (dense) and routed (gathered)
  k_gemm_h<<<dim3(HD / TN, TOK / TM, ES), 256, 0, stream>>>(
      Xs, W1ts, b1_s, Hs, nullptr, nullptr, 0);
  k_gemm_h<<<dim3(HD / TN, CAP / TM, ER), 256, 0, stream>>>(
      Xr, W1tr, b1_r, Hr, tokl, counts, 1);

  // GEMM2: shared fuses residual + both experts and fully initializes d_out
  k_out_shared<<<dim3(CD / TN, TOK / TM), 256, 0, stream>>>(Hs, W2ts, b2_s, u, out);
  // GEMM2: routed per-slot outputs, then deterministic gated combine
  k_out_routed<<<dim3(CD / TN, CAP / TM, ER), 256, 0, stream>>>(Hr, W2tr, b2_r, counts, Or);
  k_finalize<<<TOK, 256, 0, stream>>>(Or, slotof, gateof, out);
}
